// MambaLikeLM_8134668059076
// MI455X (gfx1250) — matrix operand order repaired and checked by reading, not compiled, not run on hardware
//
#include <hip/hip_runtime.h>
#include <hip/hip_bf16.h>
#include <math.h>

// ---------------- problem constants ----------------
#define BATCH   2
#define SEQLEN  2048
#define M_ROWS  4096          // BATCH*SEQLEN
#define DMODEL  1024
#define DINNER  2048
#define DSSM    2016
#define DSTATE  16
#define NHEADS  8
#define HEADP   252
#define DCAT    4064          // DINNER + DSSM

// GEMM1 (u @ W_in), with unused columns [4096,6144) of the 8200-wide output dropped.
// Remapped columns: [0,4096) = z0|x0, [4096,6144) = xBC, [6144,6152) = dt_raw.
#define KT1     32            // K = 1024
#define NT1     388           // ceil(6152/16)=385 -> pad to multiple of 4
#define LDZ     6208          // NT1*16
#define NREAL1  6152
// GEMM2 (y @ gate_w)
#define KT2     63            // K = 2016
#define NT2     128           // ceil(2016/16)=126 -> pad to 128
#define LDG     2048
// GEMM3 (concat @ W_out)
#define KT3     127           // K = 4064
#define NT3     64            // N = 1024

typedef __attribute__((ext_vector_type(16))) __bf16 v16bf;
typedef __attribute__((ext_vector_type(8)))  float  v8f;

__device__ __forceinline__ unsigned short f2bf(float f) {
    unsigned int u = __float_as_uint(f);
    unsigned int r = (u + 0x7FFFu + ((u >> 16) & 1u)) >> 16;  // round-to-nearest-even
    return (unsigned short)r;
}
__device__ __forceinline__ float sigmoidf_(float x) { return 1.0f / (1.0f + __expf(-x)); }

#define WMMA_BF16(A, B, Cacc) \
    __builtin_amdgcn_wmma_f32_16x16x32_bf16(false, (A), false, (B), (short)0, (Cacc), false, false)

// ---------------------------------------------------------------------------
// Pack f32 row-major A (M x K) into per-wave WMMA A fragments (bf16).
// Layout: [mtile][ktile][lane(32)][16 bf16], each lane = 32 contiguous bytes.
// A 16-bit 16x32 fragment: lane m = lane&15, half = lane>>4,
//   element e = 2v+p -> k = (v<4?0:16) + half*8 + (v&3)*2 + p   (ISA 7.12.2)
// ---------------------------------------------------------------------------
__global__ void k_pack_a(const float* __restrict__ src, unsigned short* __restrict__ dst,
                         int ktiles, int ldsrc) {
    size_t idx = (size_t)blockIdx.x * blockDim.x + threadIdx.x;
    int p    = (int)(idx & 1);
    int v    = (int)((idx >> 1) & 7);
    int lane = (int)((idx >> 4) & 31);
    size_t tp = idx >> 9;
    int kt = (int)(tp % (size_t)ktiles);
    int mt = (int)(tp / (size_t)ktiles);
    int m    = mt * 16 + (lane & 15);
    int half = lane >> 4;
    int k    = kt * 32 + ((v < 4) ? 0 : 16) + half * 8 + (v & 3) * 2 + p;
    dst[idx] = f2bf(src[(size_t)m * ldsrc + k]);
}

// ---------------------------------------------------------------------------
// Pack f32 row-major B (K x N) into WMMA B fragments (bf16), zero-padding
// out-of-range N. Layout: [ktile][ntile][lane(32)][16 bf16].
// B 16-bit 32x16 fragment: lane n = lane&15, half = lane>>4, k = (v<4?0:16) + half*8 + (v&3)*2 + p.
// remap!=0 drops source columns [4096,6144) (W_in unused slice).
// ---------------------------------------------------------------------------
__global__ void k_pack_b(const float* __restrict__ src, unsigned short* __restrict__ dst,
                         int ktiles, int ntiles, int Kreal, int Nreal, int ldsrc, int remap) {
    size_t idx = (size_t)blockIdx.x * blockDim.x + threadIdx.x;
    int p    = (int)(idx & 1);
    int v    = (int)((idx >> 1) & 7);
    int lane = (int)((idx >> 4) & 31);
    size_t tp = idx >> 9;
    int nt = (int)(tp % (size_t)ntiles);
    int kt = (int)(tp / (size_t)ntiles);
    int n    = nt * 16 + (lane & 15);
    int half = lane >> 4;
    int k    = kt * 32 + ((v < 4) ? 0 : 16) + half * 8 + (v & 3) * 2 + p;
    if (remap && n >= 4096) n += 2048;
    float val = (k < Kreal && n < Nreal) ? src[(size_t)k * ldsrc + n] : 0.0f;
    dst[idx] = f2bf(val);
}

// ---------------------------------------------------------------------------
// bf16 WMMA GEMM, 16x64 output per wave (4 accumulators share one A fragment),
// register double-buffered: iteration k's WMMAs consume fragments loaded at
// iteration k-1 while k+1's loads are in flight (partial s_wait_loadcnt).
// Next-iteration loads are unconditional at kt+1; the final one-past-the-end
// read stays inside the workspace by construction and is discarded.
// grid = (ntiles/4, mtiles), block = 32 (one wave; EXEC all-ones everywhere).
// ~120 VGPRs: 32 acc + 2x40 fragments + addressing -> no scratch spills.
// ---------------------------------------------------------------------------
__global__ void k_gemm(const unsigned short* __restrict__ Ap,
                       const unsigned short* __restrict__ Bp,
                       float* __restrict__ C,
                       int ktiles, int ntiles, int ldc, int Nreal) {
    int lane = threadIdx.x;
    int mt   = blockIdx.y;
    int n0   = blockIdx.x * 4;
    const size_t bstride = (size_t)ntiles * 512;

    const unsigned short* baseA = Ap + ((size_t)mt * ktiles) * 512 + (size_t)lane * 16;
    const unsigned short* baseB = Bp + (size_t)n0 * 512 + (size_t)lane * 16;

    // prologue: fragments for kt = 0
    v16bf a  = *(const v16bf*)(baseA);
    v16bf b0 = *(const v16bf*)(baseB);
    v16bf b1 = *(const v16bf*)(baseB + 512);
    v16bf b2 = *(const v16bf*)(baseB + 1024);
    v16bf b3 = *(const v16bf*)(baseB + 1536);

    v8f acc0 = {}, acc1 = {}, acc2 = {}, acc3 = {};

    for (int kt = 0; kt < ktiles; ++kt) {
        // issue next-iteration loads before this iteration's matrix ops
        const unsigned short* pa = baseA + (size_t)(kt + 1) * 512;
        const unsigned short* pb = baseB + (size_t)(kt + 1) * bstride;
        v16bf na  = *(const v16bf*)(pa);
        v16bf nb0 = *(const v16bf*)(pb);
        v16bf nb1 = *(const v16bf*)(pb + 512);
        v16bf nb2 = *(const v16bf*)(pb + 1024);
        v16bf nb3 = *(const v16bf*)(pb + 1536);
        __builtin_prefetch(pb + bstride, 0, 1);   // keep L2 a step ahead

        acc0 = WMMA_BF16(a, b0, acc0);
        acc1 = WMMA_BF16(a, b1, acc1);
        acc2 = WMMA_BF16(a, b2, acc2);
        acc3 = WMMA_BF16(a, b3, acc3);

        a = na; b0 = nb0; b1 = nb1; b2 = nb2; b3 = nb3;
    }

    // C/D layout: n = lane&15, m = v + 8*(lane>>4)
    int nlane = lane & 15;
    int mbase = mt * 16 + (lane >> 4) * 8;
    #pragma unroll
    for (int v = 0; v < 8; ++v) {
        size_t row = (size_t)(mbase + v) * ldc;
        int c0 = (n0 + 0) * 16 + nlane; if (c0 < Nreal) C[row + c0] = acc0[v];
        int c1 = (n0 + 1) * 16 + nlane; if (c1 < Nreal) C[row + c1] = acc1[v];
        int c2 = (n0 + 2) * 16 + nlane; if (c2 < Nreal) C[row + c2] = acc2[v];
        int c3 = (n0 + 3) * 16 + nlane; if (c3 < Nreal) C[row + c3] = acc3[v];
    }
}

// dt = softplus(dt_raw + dt_bias); dt_raw lives at remapped cols [6144,6152) of zx
__global__ void k_dt(const float* __restrict__ zx, const float* __restrict__ dt_bias,
                     float* __restrict__ dt) {
    size_t idx = (size_t)blockIdx.x * blockDim.x + threadIdx.x;  // M_ROWS*8
    int h = (int)(idx & 7);
    size_t r = idx >> 3;
    float x = zx[r * LDZ + 6144 + h] + dt_bias[h];
    dt[idx] = (x > 20.0f) ? x : log1pf(__expf(x));
}

// causal depthwise conv (width 4) + bias + SiLU over xBC (remapped cols [4096,6144))
__global__ void k_conv(const float* __restrict__ zx, const float* __restrict__ cw,
                       const float* __restrict__ cb, float* __restrict__ out) {
    size_t idx = (size_t)blockIdx.x * blockDim.x + threadIdx.x;  // M_ROWS*2048
    int c = (int)(idx & 2047);
    size_t r = idx >> 11;
    int t = (int)(r & (SEQLEN - 1));
    float acc = cb[c];
    const float* base = zx + r * LDZ + 4096 + c;
    #pragma unroll
    for (int k = 0; k < 4; ++k) {
        int tt = t + k - 3;
        if (tt >= 0) acc += base[(ptrdiff_t)(k - 3) * LDZ] * cw[c * 4 + k];
    }
    out[idx] = acc * sigmoidf_(acc);
}

// sequential SSM scan: one block per (batch, head); thread owns p, 16 states in regs
__global__ void k_scan(const float* __restrict__ xbc, const float* __restrict__ dt,
                       const float* __restrict__ A_log, const float* __restrict__ Dp,
                       float* __restrict__ y) {
    int b  = blockIdx.x >> 3;
    int hd = blockIdx.x & 7;
    int p  = threadIdx.x;
    bool act = p < HEADP;
    float Ah = -__expf(A_log[hd]);
    float Dh = Dp[hd];
    float h[DSTATE];
    #pragma unroll
    for (int n = 0; n < DSTATE; ++n) h[n] = 0.0f;
    __shared__ float Bs[DSTATE];
    __shared__ float Cs[DSTATE];
    __shared__ float sdt;
    for (int t = 0; t < SEQLEN; ++t) {
        size_t row = (size_t)(b * SEQLEN + t);
        if (p < 16)       Bs[p]      = xbc[row * 2048 + 2016 + p];
        else if (p < 32)  Cs[p - 16] = xbc[row * 2048 + 2032 + (p - 16)];
        else if (p == 32) sdt        = dt[row * 8 + hd];
        __syncthreads();
        if (act) {
            float xt = xbc[row * 2048 + hd * HEADP + p];
            float d  = sdt;
            float dA = __expf(d * Ah);
            float dx = d * xt;
            float acc = Dh * xt;
            #pragma unroll
            for (int n = 0; n < DSTATE; ++n) {
                h[n] = dA * h[n] + dx * Bs[n];
                acc += h[n] * Cs[n];
            }
            y[row * DSSM + hd * HEADP + p] = acc;
        }
        __syncthreads();
    }
}

// in-place LayerNorm over DSSM
__global__ void k_ln(float* __restrict__ y, const float* __restrict__ g,
                     const float* __restrict__ bta) {
    __shared__ float r1[256];
    __shared__ float r2[256];
    size_t row = blockIdx.x;
    float s = 0.0f, s2 = 0.0f;
    for (int c = threadIdx.x; c < DSSM; c += 256) {
        float v = y[row * DSSM + c];
        s += v; s2 += v * v;
    }
    r1[threadIdx.x] = s; r2[threadIdx.x] = s2;
    __syncthreads();
    for (int st = 128; st > 0; st >>= 1) {
        if (threadIdx.x < st) { r1[threadIdx.x] += r1[threadIdx.x + st]; r2[threadIdx.x] += r2[threadIdx.x + st]; }
        __syncthreads();
    }
    float mu   = r1[0] / (float)DSSM;
    float var  = r2[0] / (float)DSSM - mu * mu;
    float rstd = rsqrtf(var + 1e-5f);
    for (int c = threadIdx.x; c < DSSM; c += 256) {
        float v = y[row * DSSM + c];
        y[row * DSSM + c] = (v - mu) * rstd * g[c] + bta[c];
    }
}

// concat cols [0,2048) = silu(z0)*x0
__global__ void k_mlp(const float* __restrict__ zx, float* __restrict__ cat) {
    size_t idx = (size_t)blockIdx.x * blockDim.x + threadIdx.x;  // M_ROWS*2048
    int c = (int)(idx & 2047);
    size_t r = idx >> 11;
    float z  = zx[r * LDZ + c];
    float x0 = zx[r * LDZ + 2048 + c];
    cat[r * DCAT + c] = z * sigmoidf_(z) * x0;
}

// concat cols [2048,4064) = sigmoid(glog + gate_b) * y_ln
__global__ void k_gatecat(const float* __restrict__ glog, const float* __restrict__ gb,
                          const float* __restrict__ y, float* __restrict__ cat) {
    size_t idx = (size_t)blockIdx.x * blockDim.x + threadIdx.x;  // M_ROWS*DSSM
    int c = (int)(idx % DSSM);
    size_t r = idx / DSSM;
    float g = sigmoidf_(glog[r * LDG + c] + gb[c]);
    cat[r * DCAT + 2048 + c] = g * y[idx];
}

extern "C" void kernel_launch(void* const* d_in, const int* in_sizes, int n_in,
                              void* d_out, int out_size, void* d_ws, size_t ws_size,
                              hipStream_t stream) {
    (void)in_sizes; (void)n_in; (void)out_size; (void)ws_size;
    const float* u      = (const float*)d_in[0];
    const float* W_in   = (const float*)d_in[1];
    const float* conv_w = (const float*)d_in[2];
    const float* conv_b = (const float*)d_in[3];
    const float* dt_b   = (const float*)d_in[4];
    const float* A_log  = (const float*)d_in[5];
    const float* Dp     = (const float*)d_in[6];
    const float* ln_g   = (const float*)d_in[7];
    const float* ln_b   = (const float*)d_in[8];
    const float* gate_w = (const float*)d_in[9];
    const float* gate_b = (const float*)d_in[10];
    const float* W_out  = (const float*)d_in[11];
    float* out = (float*)d_out;

    // -------- workspace layout (lifetime-overlapped regions, 256B aligned) --------
    constexpr size_t SZ_PAU  = (size_t)M_ROWS * 1024 * 2;      // packed u (bf16)
    constexpr size_t SZ_PAY  = (size_t)M_ROWS * DSSM * 2;      // packed y_ln
    constexpr size_t SZ_PBG  = (size_t)KT2 * NT2 * 512 * 2;    // packed gate_w
    constexpr size_t SZ_REGA = SZ_PAY + SZ_PBG;                // largest simultaneous pair
    constexpr size_t OFF_ZX  = SZ_REGA;
    constexpr size_t SZ_ZX   = (size_t)M_ROWS * LDZ * 4;
    constexpr size_t OFF_DT  = OFF_ZX + SZ_ZX;
    constexpr size_t SZ_DT   = (size_t)M_ROWS * 8 * 4;
    constexpr size_t OFF_XBC = OFF_DT + SZ_DT;
    constexpr size_t SZ_XBC  = (size_t)M_ROWS * 2048 * 4;
    constexpr size_t OFF_Y   = OFF_XBC + SZ_XBC;
    constexpr size_t SZ_Y    = (size_t)M_ROWS * DSSM * 4;
    constexpr size_t OFF_G   = OFF_Y + SZ_Y;                   // gate logits, later packed concat
    constexpr size_t SZ_G    = (size_t)M_ROWS * LDG * 4;
    constexpr size_t OFF_CAT = OFF_G + SZ_G;

    char* ws = (char*)d_ws;
    unsigned short* pAu = (unsigned short*)(ws);               // region A, phase 1
    unsigned short* pBW = (unsigned short*)(ws + SZ_PAU);
    unsigned short* pAy = (unsigned short*)(ws);               // region A, phase 2
    unsigned short* pBg = (unsigned short*)(ws + SZ_PAY);
    unsigned short* pBo = (unsigned short*)(ws);               // region A, phase 3
    float* zx   = (float*)(ws + OFF_ZX);
    float* dt   = (float*)(ws + OFF_DT);
    float* xbc  = (float*)(ws + OFF_XBC);
    float* y    = (float*)(ws + OFF_Y);
    float* glog = (float*)(ws + OFF_G);
    unsigned short* pAc = (unsigned short*)(ws + OFF_G);       // reuses glog region
    float* cat  = (float*)(ws + OFF_CAT);

    // 1) pack A = u (4096x1024), B = W_in (1024x8200, column-remapped, N-padded)
    k_pack_a<<<(M_ROWS * 1024) / 256, 256, 0, stream>>>(u, pAu, KT1, DMODEL);
    k_pack_b<<<(KT1 * NT1 * 512) / 256, 256, 0, stream>>>(W_in, pBW, KT1, NT1, 1024, 8200, 8200, 1);
    // 2) GEMM1 -> zx (remapped, ld = LDZ)
    {
        dim3 g(NT1 / 4, M_ROWS / 16);
        k_gemm<<<g, 32, 0, stream>>>(pAu, pBW, zx, KT1, NT1, LDZ, NREAL1);
    }
    // 3) dt = softplus(dt_raw + bias)
    k_dt<<<(M_ROWS * 8) / 256, 256, 0, stream>>>(zx, dt_b, dt);
    // 4) conv + SiLU -> xbc
    k_conv<<<(M_ROWS * 2048) / 256, 256, 0, stream>>>(zx, conv_w, conv_b, xbc);
    // 5) sequential SSM scan -> y
    k_scan<<<BATCH * NHEADS, 256, 0, stream>>>(xbc, dt, A_log, Dp, y);
    // 6) LayerNorm (in place)
    k_ln<<<M_ROWS, 256, 0, stream>>>(y, ln_g, ln_b);
    // 7) pack A = y_ln, B = gate_w; GEMM2 -> glog
    k_pack_a<<<(M_ROWS * DSSM) / 256, 256, 0, stream>>>(y, pAy, KT2, DSSM);
    k_pack_b<<<(KT2 * NT2 * 512) / 256, 256, 0, stream>>>(gate_w, pBg, KT2, NT2, DSSM, DSSM, DSSM, 0);
    {
        dim3 g(NT2 / 4, M_ROWS / 16);
        k_gemm<<<g, 32, 0, stream>>>(pAy, pBg, glog, KT2, NT2, LDG, DSSM);
    }
    // 8) build concat = [silu(z0)*x0 | sigmoid(glog)*y_ln]
    k_mlp<<<(M_ROWS * 2048) / 256, 256, 0, stream>>>(zx, cat);
    k_gatecat<<<(M_ROWS * DSSM) / 256, 256, 0, stream>>>(glog, gate_b, y, cat);
    // 9) pack A = concat (reuses glog region), B = W_out; GEMM3 -> d_out
    k_pack_a<<<(M_ROWS * DCAT) / 256, 256, 0, stream>>>(cat, pAc, KT3, DCAT);
    k_pack_b<<<(KT3 * NT3 * 512) / 256, 256, 0, stream>>>(W_out, pBo, KT3, NT3, DCAT, 1024, 1024, 0);
    {
        dim3 g(NT3 / 4, M_ROWS / 16);
        k_gemm<<<g, 32, 0, stream>>>(pAc, pBo, out, KT3, NT3, 1024, 1024);
    }
}
